// AttentionEmbedding_51238959841627
// MI455X (gfx1250) — matrix operand order repaired and checked by reading, not compiled, not run on hardware
//
#include <hip/hip_runtime.h>

// ---------------------------------------------------------------------------
// TGN temporal attention, fused per 16-row tile, f16 WMMA on gfx1250 (CDNA5).
// Math restructuring (exact linear-algebra identities, fp reassociation only):
//   scores = (Wk_h^T q_h) . keyv          (fold Wk into q: 16x2 rows, not 160)
//   out    = (attn @ keyv) @ Wv^T         (fold attn before Wv: 16 rows, not 160)
// GEMM engine: v_wmma_f32_16x16x32_f16, A from LDS, B from f16 weights in L2.
// 512 threads = 16 wave32 per block; one block per WGP (213KB LDS).
// ---------------------------------------------------------------------------

typedef _Float16 h16;
typedef __attribute__((ext_vector_type(16))) _Float16 v16h;
typedef __attribute__((ext_vector_type(8)))  _Float16 v8h;
typedef __attribute__((ext_vector_type(4)))  _Float16 v4h;
typedef __attribute__((ext_vector_type(8)))  float    v8f;

#define DF   128      // node/edge feature dim
#define TF   128      // time encoding dim
#define QD   256      // query dim
#define KD   384      // key dim
#define KNB  10       // neighbors per row
#define TM   16       // rows per tile (one WMMA M-tile)
#define NTHR 512      // 16 wave32 per workgroup

// ---- dynamic LDS layout (bytes), all chunks 16B aligned -------------------
#define LDS_QRY   0                     // [16][256] h16 query (src||tenc0)   8192
#define LDS_KV    (LDS_QRY + 8192)      // [160][384] h16 keyv              122880
#define LDS_QPR   (LDS_KV + 122880)     // [16][256] h16 projected q          8192
#define LDS_QK    (LDS_QPR + 8192)      // [16][768] h16 Wk-folded q         24576
#define LDS_CTX   (LDS_QK + 24576)      // [16][768] h16 attn-weighted ctx   24576
#define LDS_MHA   (LDS_CTX + 24576)     // [16][256] h16 ctx @ Wv^T           8192
#define LDS_CAT   (LDS_MHA + 8192)      // [16][384] h16 [out||src]          12288
#define LDS_H1    (LDS_CAT + 12288)     // [16][128] h16 relu hidden          4096
#define LDS_SC    (LDS_H1 + 4096)       // [16][2][10] f32 scores             1280
#define LDS_AT    (LDS_SC + 1280)       // [16][2][10] f32 attn               1280
#define LDS_QBK   (LDS_AT + 1280)       // [16][2] f32 q.bk                    128
#define LDS_MSK   (LDS_QBK + 128)       // [160] int mask                      640
#define LDS_INV   (LDS_MSK + 640)       // [16] int invalid-row                 64
#define LDS_NBR   (LDS_INV + 64)        // [160] int neighbor id               640
#define LDS_EID   (LDS_NBR + 640)       // [160] int edge id                   640
#define LDS_DT    (LDS_EID + 640)       // [160] f32 ts - edge_time            640
#define LDS_TOTAL (LDS_DT + 640)        // = 218304 B  (< 320KB/WGP)

// ---- WMMA fragment helpers (ISA 7.12.2 layouts) ---------------------------
// A 16x32 f16: lanes 0-15 -> row m=lane, K k0+0..7 then k0+16..23
//              lanes 16-31 -> row m=lane-16, K k0+8..15 then k0+24..31
__device__ __forceinline__ v16h frag_a(const h16* base, int stride, int k0) {
  int lane = threadIdx.x & 31;
  int m  = lane & 15;
  int ko = (lane & 16) ? 8 : 0;
  const h16* p = base + m * stride + k0 + ko;
  union { v16h v; v8h h[2]; } u;
  u.h[0] = *(const v8h*)(p);
  u.h[1] = *(const v8h*)(p + 16);
  return u.v;
}

// B 32x16 f16 stored as W[n][k] row-major: lanes 0-15 -> col n, K k0+0..7 then k0+16..23
//                                          lanes 16-31 -> col n, K k0+8..15 then k0+24..31
__device__ __forceinline__ v16h frag_b(const h16* base, int stride, int n0, int k0) {
  int lane = threadIdx.x & 31;
  int n  = lane & 15;
  int ko = (lane & 16) ? 8 : 0;
  const h16* p = base + (size_t)(n0 + n) * stride + k0 + ko;
  union { v16h v; v8h h[2]; } u;
  u.h[0] = *(const v8h*)(p);
  u.h[1] = *(const v8h*)(p + 16);
  return u.v;
}

__device__ __forceinline__ v8f wmma32(v16h a, v16h b, v8f c) {
  return __builtin_amdgcn_wmma_f32_16x16x32_f16(false, a, false, b, (short)0, c,
                                                false, false);
}

__device__ __forceinline__ v4h pack4(float x, float y, float z, float w) {
  v4h h; h[0] = (h16)x; h[1] = (h16)y; h[2] = (h16)z; h[3] = (h16)w;
  return h;
}

// ---------------------------------------------------------------------------
__global__ __launch_bounds__(NTHR) void tgn_attn_kernel(
    int mode,                                   // 0: L1 src, 1: L1 neigh, 2: L2
    const float* __restrict__ node_feat, const float* __restrict__ memv,
    const float* __restrict__ edge_feat,
    const float* __restrict__ time_w, const float* __restrict__ time_b,
    const h16*  __restrict__ tenc0H,
    const int*  __restrict__ src_nodes,
    const int*  __restrict__ neighbors1, const int* __restrict__ edge_idx1,
    const float* __restrict__ edge_times1,
    const int*  __restrict__ neighbors2, const int* __restrict__ edge_idx2,
    const float* __restrict__ edge_times2,
    const float* __restrict__ timestamps,
    const float* __restrict__ srcPrev,          // mode 2: src_l1  [M,128]
    const float* __restrict__ neighPrev,        // mode 2: neigh_l1 [M*10,128]
    const h16* __restrict__ WqH,  const h16* __restrict__ WkTH,
    const h16* __restrict__ WvH,  const h16* __restrict__ WoH,
    const h16* __restrict__ W1H,  const h16* __restrict__ W2H,
    const float* __restrict__ bq, const float* __restrict__ bk,
    const float* __restrict__ bv, const float* __restrict__ bo,
    const float* __restrict__ b1, const float* __restrict__ b2,
    float* __restrict__ outp)
{
  extern __shared__ char smem[];
  h16*  sQry = (h16*)(smem + LDS_QRY);
  h16*  sKV  = (h16*)(smem + LDS_KV);
  h16*  sQpr = (h16*)(smem + LDS_QPR);
  h16*  sQK  = (h16*)(smem + LDS_QK);
  h16*  sCtx = (h16*)(smem + LDS_CTX);
  h16*  sMha = (h16*)(smem + LDS_MHA);
  h16*  sCat = (h16*)(smem + LDS_CAT);
  h16*  sH1  = (h16*)(smem + LDS_H1);
  float* sSc  = (float*)(smem + LDS_SC);
  float* sAt  = (float*)(smem + LDS_AT);
  float* sQbk = (float*)(smem + LDS_QBK);
  int*   sMsk = (int*)(smem + LDS_MSK);
  int*   sInv = (int*)(smem + LDS_INV);
  int*   sNbr = (int*)(smem + LDS_NBR);
  int*   sEid = (int*)(smem + LDS_EID);
  float* sDt  = (float*)(smem + LDS_DT);

  const int tid = threadIdx.x;
  const int wv  = tid >> 5;
  const int m0  = blockIdx.x * TM;

  // ---------- warm L2/WGP$ with this pass's f16 weight panels ----------
  {
    const h16* wptr[6] = {WqH, WkTH, WvH, WoH, W1H, W2H};
    const int  wsz[6]  = {QD * QD, KD * QD, QD * KD, QD * QD, DF * KD, DF * DF};
    #pragma unroll
    for (int w = 0; w < 6; ++w)
      for (int i = tid * 64; i < wsz[w]; i += NTHR * 64)
        __builtin_prefetch(wptr[w] + i, 0, 3);   // -> global_prefetch_b8
  }

  // ---------- phase 0a: per-neighbor metadata ----------
  for (int r = tid; r < TM * KNB; r += NTHR) {
    int m = m0 + r / KNB;
    int e = m * KNB + (r % KNB);
    int nb   = (mode == 1) ? neighbors2[e]  : neighbors1[e];
    int eid  = (mode == 1) ? edge_idx2[e]   : edge_idx1[e];
    float ts = (mode == 1) ? timestamps[m / KNB] : timestamps[m];
    float et = (mode == 1) ? edge_times2[e] : edge_times1[e];
    sNbr[r] = nb;
    sEid[r] = eid;
    sDt[r]  = ts - et;
    sMsk[r] = (nb == 0) ? 1 : 0;
  }
  __syncthreads();
  if (tid < TM) {                     // invalid rows: unmask slot 0 (ref semantics)
    int inv = 1;
    #pragma unroll
    for (int k = 0; k < KNB; ++k) inv &= sMsk[tid * KNB + k];
    sInv[tid] = inv;
    if (inv) sMsk[tid * KNB] = 0;
  }

  // ---------- phase 0b: gather query (src emb || tenc(0)) + src copy ----------
  for (int idx = tid; idx < TM * DF / 4; idx += NTHR) {
    int mr = idx / (DF / 4), c4 = (idx % (DF / 4)) * 4;
    int m = m0 + mr;
    float4 v;
    if (mode == 2) {
      v = *(const float4*)(srcPrev + (size_t)m * DF + c4);
    } else {
      int s = (mode == 0) ? src_nodes[m] : neighbors1[m];
      float4 a = *(const float4*)(node_feat + (size_t)s * DF + c4);
      float4 b = *(const float4*)(memv + (size_t)s * DF + c4);
      v = make_float4(a.x + b.x, a.y + b.y, a.z + b.z, a.w + b.w);
    }
    v4h hv = pack4(v.x, v.y, v.z, v.w);
    *(v4h*)(sQry + mr * QD + c4) = hv;
    *(v4h*)(sCat + mr * KD + QD + c4) = hv;     // right half of [out || src]
  }
  for (int idx = tid; idx < TM * TF / 4; idx += NTHR) {
    int mr = idx / (TF / 4), c4 = (idx % (TF / 4)) * 4;
    *(v4h*)(sQry + mr * QD + DF + c4) = *(const v4h*)(tenc0H + c4);
  }
  __syncthreads();

  // ---------- phase 0c: gather keyv = [neigh_emb || edge_feat || tenc(dt)] ----
  for (int idx = tid; idx < TM * KNB * KD / 4; idx += NTHR) {
    int r = idx / (KD / 4), c4 = (idx % (KD / 4)) * 4;
    float4 v;
    if (c4 < DF) {
      if (mode == 2) {
        int m = m0 + r / KNB;
        v = *(const float4*)(neighPrev + ((size_t)m * KNB + (r % KNB)) * DF + c4);
      } else {
        int nb = sNbr[r];
        float4 a = *(const float4*)(node_feat + (size_t)nb * DF + c4);
        float4 b = *(const float4*)(memv + (size_t)nb * DF + c4);
        v = make_float4(a.x + b.x, a.y + b.y, a.z + b.z, a.w + b.w);
      }
    } else if (c4 < 2 * DF) {
      v = *(const float4*)(edge_feat + (size_t)sEid[r] * DF + (c4 - DF));
    } else {
      int cc = c4 - 2 * DF;
      float4 w  = *(const float4*)(time_w + cc);
      float4 bb = *(const float4*)(time_b + cc);
      float dt = sDt[r];
      v = make_float4(__cosf(dt * w.x + bb.x), __cosf(dt * w.y + bb.y),
                      __cosf(dt * w.z + bb.z), __cosf(dt * w.w + bb.w));
    }
    *(v4h*)(sKV + r * KD + c4) = pack4(v.x, v.y, v.z, v.w);
  }
  __syncthreads();

  // ---------- G1: q = Q @ Wq^T + bq   [16,256] ----------
  for (int t = wv; t < 16; t += 16) {
    int n0 = t * 16;
    v8f acc = {};
    #pragma unroll
    for (int kk = 0; kk < 8; ++kk)
      acc = wmma32(frag_a(sQry, QD, kk * 32), frag_b(WqH, QD, n0, kk * 32), acc);
    int lane = tid & 31, n = lane & 15, mb = (lane & 16) ? 8 : 0;
    float bias = bq[n0 + n];
    #pragma unroll
    for (int i = 0; i < 8; ++i)
      sQpr[(mb + i) * QD + n0 + n] = (h16)(acc[i] + bias);
  }
  __syncthreads();

  // ---------- G2: qk_h = q_h @ Wk_h   [16,2,384]  (Wk folded into q) ----------
  for (int t = wv; t < 48; t += 16) {
    int h = t / 24, n0 = (t % 24) * 16;
    v8f acc = {};
    #pragma unroll
    for (int kk = 0; kk < 4; ++kk)
      acc = wmma32(frag_a(sQpr + h * DF, QD, kk * 32),
                   frag_b(WkTH, QD, n0, h * DF + kk * 32), acc);
    int lane = tid & 31, n = lane & 15, mb = (lane & 16) ? 8 : 0;
    #pragma unroll
    for (int i = 0; i < 8; ++i)
      sQK[(mb + i) * (2 * KD) + h * KD + n0 + n] = (h16)acc[i];
  }
  __syncthreads();

  // ---------- q . bk per (m,h) ----------
  if (tid < TM * 2) {
    int m = tid >> 1, h = tid & 1;
    float s = 0.f;
    for (int j = 0; j < DF; ++j)
      s += (float)sQpr[m * QD + h * DF + j] * bk[h * DF + j];
    sQbk[tid] = s;
  }
  __syncthreads();

  // ---------- scores + softmax ----------
  for (int idx = tid; idx < TM * 2 * KNB; idx += NTHR) {
    int m = idx / (2 * KNB), rem = idx % (2 * KNB), h = rem / KNB, kn = rem % KNB;
    const h16* qp = sQK + m * (2 * KD) + h * KD;
    const h16* kp = sKV + (m * KNB + kn) * KD;
    float s = 0.f;
    for (int d = 0; d < KD; d += 8) {
      v8h qa = *(const v8h*)(qp + d);
      v8h ka = *(const v8h*)(kp + d);
      #pragma unroll
      for (int j = 0; j < 8; ++j) s += (float)qa[j] * (float)ka[j];
    }
    s = (s + sQbk[m * 2 + h]) * 0.088388347648318447f;   // 1/sqrt(128)
    if (sMsk[m * KNB + kn]) s = -1e9f;
    sSc[idx] = s;
  }
  __syncthreads();
  if (tid < TM * 2) {
    int base = tid * KNB;
    float mx = -3.4e38f;
    #pragma unroll
    for (int k = 0; k < KNB; ++k) mx = fmaxf(mx, sSc[base + k]);
    float e[KNB], sum = 0.f;
    #pragma unroll
    for (int k = 0; k < KNB; ++k) { e[k] = __expf(sSc[base + k] - mx); sum += e[k]; }
    float inv = 1.f / sum;
    #pragma unroll
    for (int k = 0; k < KNB; ++k) sAt[base + k] = e[k] * inv;
  }
  __syncthreads();

  // ---------- ctx_h[m,d] = sum_kn attn * keyv   [16,2,384] ----------
  for (int idx = tid; idx < TM * 2 * KD / 4; idx += NTHR) {
    int m = idx / (2 * KD / 4), c = idx % (2 * KD / 4);
    int h = c / (KD / 4), d4 = (c % (KD / 4)) * 4;
    const float* ap = sAt + m * 2 * KNB + h * KNB;
    float s0 = 0.f, s1 = 0.f, s2 = 0.f, s3 = 0.f;
    #pragma unroll
    for (int k = 0; k < KNB; ++k) {
      v4h kv = *(const v4h*)(sKV + (m * KNB + k) * KD + d4);
      float a = ap[k];
      s0 += a * (float)kv[0]; s1 += a * (float)kv[1];
      s2 += a * (float)kv[2]; s3 += a * (float)kv[3];
    }
    *(v4h*)(sCtx + m * (2 * KD) + h * KD + d4) = pack4(s0, s1, s2, s3);
  }
  __syncthreads();

  // ---------- G3: mha = ctx_h @ Wv_h^T + bv   [16,256] ----------
  for (int t = wv; t < 16; t += 16) {
    int h = t >> 3, nglob = h * DF + (t & 7) * 16;
    v8f acc = {};
    #pragma unroll
    for (int kk = 0; kk < 12; ++kk)
      acc = wmma32(frag_a(sCtx + h * KD, 2 * KD, kk * 32),
                   frag_b(WvH, KD, nglob, kk * 32), acc);
    int lane = tid & 31, n = lane & 15, mb = (lane & 16) ? 8 : 0;
    float bias = bv[nglob + n];                 // attn sums to 1 per head
    #pragma unroll
    for (int i = 0; i < 8; ++i)
      sMha[(mb + i) * QD + nglob + n] = (h16)(acc[i] + bias);
  }
  __syncthreads();

  // ---------- G4: out = mha @ Wo^T + bo, zero invalid rows -> sCat[:, :256] ---
  for (int t = wv; t < 16; t += 16) {
    int n0 = t * 16;
    v8f acc = {};
    #pragma unroll
    for (int kk = 0; kk < 8; ++kk)
      acc = wmma32(frag_a(sMha, QD, kk * 32), frag_b(WoH, QD, n0, kk * 32), acc);
    int lane = tid & 31, n = lane & 15, mb = (lane & 16) ? 8 : 0;
    float bias = bo[n0 + n];
    #pragma unroll
    for (int i = 0; i < 8; ++i) {
      float vv = sInv[mb + i] ? 0.f : (acc[i] + bias);
      sCat[(mb + i) * KD + n0 + n] = (h16)vv;
    }
  }
  __syncthreads();

  // ---------- G5: h1 = relu([out||src] @ W1^T + b1)   [16,128] ----------
  if (wv < 8) {
    int n0 = wv * 16;
    v8f acc = {};
    #pragma unroll
    for (int kk = 0; kk < 12; ++kk)
      acc = wmma32(frag_a(sCat, KD, kk * 32), frag_b(W1H, KD, n0, kk * 32), acc);
    int lane = tid & 31, n = lane & 15, mb = (lane & 16) ? 8 : 0;
    float bias = b1[n0 + n];
    #pragma unroll
    for (int i = 0; i < 8; ++i)
      sH1[(mb + i) * DF + n0 + n] = (h16)fmaxf(acc[i] + bias, 0.f);
  }
  __syncthreads();

  // ---------- G6: res = h1 @ W2^T + b2 -> global f32 ----------
  if (wv < 8) {
    int n0 = wv * 16;
    v8f acc = {};
    #pragma unroll
    for (int kk = 0; kk < 4; ++kk)
      acc = wmma32(frag_a(sH1, DF, kk * 32), frag_b(W2H, DF, n0, kk * 32), acc);
    int lane = tid & 31, n = lane & 15, mb = (lane & 16) ? 8 : 0;
    float bias = b2[n0 + n];
    #pragma unroll
    for (int i = 0; i < 8; ++i)
      outp[((size_t)(m0 + mb + i)) * DF + n0 + n] = acc[i] + bias;
  }
}

// ---------------------------------------------------------------------------
// prep kernels: fp32 -> fp16 weight conversion (+ Wk transpose, tenc(0))
// ---------------------------------------------------------------------------
__global__ void cvt_f32_f16_kernel(h16* dst, const float* src, int n) {
  int i = blockIdx.x * blockDim.x + threadIdx.x;
  if (i < n) dst[i] = (h16)src[i];
}

__global__ void build_wkT_kernel(h16* dst, const float* Wk) {   // [2,384,256]
  int i = blockIdx.x * blockDim.x + threadIdx.x;
  int total = 2 * KD * QD;
  if (i < total) {
    int a = i / (KD * QD), r = i % (KD * QD);
    int d = r / QD, j = r % QD;                // dst[a][d][j] = Wk[a][j][d]
    dst[i] = (h16)Wk[(size_t)a * QD * KD + (size_t)j * KD + d];
  }
}

__global__ void build_tenc0_kernel(h16* dst, const float* time_b) {
  int i = threadIdx.x;
  if (i < TF) dst[i] = (h16)__cosf(time_b[i]);
}

// ---------------------------------------------------------------------------
extern "C" void kernel_launch(void* const* d_in, const int* in_sizes, int n_in,
                              void* d_out, int out_size, void* d_ws, size_t ws_size,
                              hipStream_t stream) {
  (void)in_sizes; (void)n_in; (void)out_size; (void)ws_size;
  const float* node_feat   = (const float*)d_in[0];
  const float* memv        = (const float*)d_in[1];
  const float* edge_feat   = (const float*)d_in[2];
  const float* time_w      = (const float*)d_in[3];
  const float* time_b      = (const float*)d_in[4];
  const float* Wq          = (const float*)d_in[5];
  const float* bq          = (const float*)d_in[6];
  const float* Wk          = (const float*)d_in[7];
  const float* bk          = (const float*)d_in[8];
  const float* Wv          = (const float*)d_in[9];
  const float* bv          = (const float*)d_in[10];
  const float* Wo          = (const float*)d_in[11];
  const float* bo          = (const float*)d_in[12];
  const float* W1          = (const float*)d_in[13];
  const float* b1          = (const float*)d_in[14];
  const float* W2          = (const float*)d_in[15];
  const float* b2          = (const float*)d_in[16];
  const float* timestamps  = (const float*)d_in[17];
  const int*   src_nodes   = (const int*)d_in[18];
  const int*   neighbors1  = (const int*)d_in[19];
  const int*   edge_idx1   = (const int*)d_in[20];
  const float* edge_times1 = (const float*)d_in[21];
  const int*   neighbors2  = (const int*)d_in[22];
  const int*   edge_idx2   = (const int*)d_in[23];
  const float* edge_times2 = (const float*)d_in[24];

  const int B = 8192;
  char* ws = (char*)d_ws;
  size_t off = 0;
  auto take = [&](size_t bytes) -> char* {
    char* p = ws + off;
    off = (off + bytes + 255) & ~(size_t)255;
    return p;
  };
  h16* WqH    = (h16*)take((size_t)2 * QD * QD * sizeof(h16));
  h16* WkTH   = (h16*)take((size_t)2 * KD * QD * sizeof(h16));
  h16* WvH    = (h16*)take((size_t)2 * QD * KD * sizeof(h16));
  h16* WoH    = (h16*)take((size_t)2 * QD * QD * sizeof(h16));
  h16* W1H    = (h16*)take((size_t)2 * DF * KD * sizeof(h16));
  h16* W2H    = (h16*)take((size_t)2 * DF * DF * sizeof(h16));
  h16* tenc0H = (h16*)take((size_t)TF * sizeof(h16));
  float* src_l1   = (float*)take((size_t)B * DF * sizeof(float));
  float* neigh_l1 = (float*)take((size_t)B * KNB * DF * sizeof(float));

  auto cvt = [&](h16* dst, const float* src, int n) {
    cvt_f32_f16_kernel<<<(n + 255) / 256, 256, 0, stream>>>(dst, src, n);
  };
  cvt(WqH, Wq, 2 * QD * QD);
  cvt(WvH, Wv, 2 * QD * KD);
  cvt(WoH, Wo, 2 * QD * QD);
  cvt(W1H, W1, 2 * DF * KD);
  cvt(W2H, W2, 2 * DF * DF);
  build_wkT_kernel<<<(2 * KD * QD + 255) / 256, 256, 0, stream>>>(WkTH, Wk);
  build_tenc0_kernel<<<1, 128, 0, stream>>>(tenc0H, time_b);

  static_assert(LDS_TOTAL <= 320 * 1024, "LDS budget");
  (void)hipFuncSetAttribute((const void*)tgn_attn_kernel,
                            hipFuncAttributeMaxDynamicSharedMemorySize, LDS_TOTAL);

  // pass 0: layer-1 embedding of src nodes (params[0]) -> src_l1
  tgn_attn_kernel<<<B / TM, NTHR, LDS_TOTAL, stream>>>(
      0, node_feat, memv, edge_feat, time_w, time_b, tenc0H,
      src_nodes, neighbors1, edge_idx1, edge_times1,
      neighbors2, edge_idx2, edge_times2, timestamps,
      nullptr, nullptr,
      WqH, WkTH, WvH, WoH, W1H, W2H,
      bq, bk, bv, bo, b1, b2,
      src_l1);

  // pass 1: layer-1 embedding of flattened 1-hop neighbors (params[0]) -> neigh_l1
  tgn_attn_kernel<<<(B * KNB) / TM, NTHR, LDS_TOTAL, stream>>>(
      1, node_feat, memv, edge_feat, time_w, time_b, tenc0H,
      src_nodes, neighbors1, edge_idx1, edge_times1,
      neighbors2, edge_idx2, edge_times2, timestamps,
      nullptr, nullptr,
      WqH, WkTH, WvH, WoH, W1H, W2H,
      bq, bk, bv, bo, b1, b2,
      neigh_l1);

  // pass 2: layer-2 aggregation (params[1]) -> d_out
  tgn_attn_kernel<<<B / TM, NTHR, LDS_TOTAL, stream>>>(
      2, node_feat, memv, edge_feat, time_w, time_b, tenc0H,
      src_nodes, neighbors1, edge_idx1, edge_times1,
      neighbors2, edge_idx2, edge_times2, timestamps,
      src_l1, neigh_l1,
      WqH + QD * QD, WkTH + KD * QD, WvH + QD * KD, WoH + QD * QD,
      W1H + DF * KD, W2H + DF * DF,
      bq + QD, bk + QD, bv + QD, bo + QD, b1 + DF, b2 + DF,
      (float*)d_out);
}
